// TransformerBlock_26276609917693
// MI455X (gfx1250) — matrix operand order repaired and checked by reading, not compiled, not run on hardware
//
#include <hip/hip_runtime.h>
#include <math.h>

// Problem constants (compile-time known from the reference)
#define BB   4
#define SS   2048
#define EE   1024
#define CSZ  16
#define NCC  8
#define HH   16
#define DD   1040        // E + CS
#define DHH  65          // D / H
#define DHP  96          // DH padded to multiple of 32
#define DP   1056        // D padded to multiple of 32 (K dim for q/v/o GEMMs)
#define DFF  4096
#define K2   8352        // NC*D + NC (=8328) padded to multiple of 32
#define NTOK (BB*SS)     // 8192

typedef __attribute__((ext_vector_type(16))) __bf16          v16bf;
typedef __attribute__((ext_vector_type(8)))  float           v8f;
typedef __attribute__((ext_vector_type(8)))  unsigned short  ushort8;
typedef __attribute__((ext_vector_type(4)))  int             v4i;

#define GLOBAL_AS __attribute__((address_space(1)))
#define LDS_AS    __attribute__((address_space(3)))

#if __has_builtin(__builtin_amdgcn_global_load_async_to_lds_b128) && \
    __has_builtin(__builtin_amdgcn_s_wait_asynccnt)
#define USE_ASYNC 1
#else
#define USE_ASYNC 0
#endif

static __device__ __forceinline__ unsigned short f2bf(float f) {
  union { float f; unsigned u; } v; v.f = f;
  unsigned u = v.u;
  u = u + 0x7FFFu + ((u >> 16) & 1u);   // round-to-nearest-even
  return (unsigned short)(u >> 16);
}
static __device__ __forceinline__ float bf2f(unsigned short h) {
  union { unsigned u; float f; } v; v.u = ((unsigned)h) << 16;
  return v.f;
}
// Assemble a 16-half WMMA operand from two 16-byte chunks.
static __device__ __forceinline__ v16bf load_frag(const unsigned short* p0,
                                                  const unsigned short* p1) {
  union { ushort8 u[2]; v16bf v; } t;
  t.u[0] = *(const ushort8*)p0;
  t.u[1] = *(const ushort8*)p1;
  return t.v;
}
static __device__ __forceinline__ v8f wmma_bf16(v16bf a, v16bf b, v8f c) {
  return __builtin_amdgcn_wmma_f32_16x16x32_bf16(false, a, false, b, (short)0, c,
                                                 false, false);
}
static __device__ __forceinline__ v8f zero8() {
  v8f z;
#pragma unroll
  for (int i = 0; i < 8; i++) z[i] = 0.f;
  return z;
}
static __device__ __forceinline__ v16bf ones16() {   // bf16 1.0 = 0x3F80
  union { ushort8 u[2]; v16bf v; } t;
#pragma unroll
  for (int i = 0; i < 8; i++) { t.u[0][i] = 0x3F80; t.u[1][i] = 0x3F80; }
  return t.v;
}

// 16-byte global -> LDS copy: async DMA (ASYNCcnt) when available.
static __device__ __forceinline__ void cp16(unsigned short* lds,
                                            const unsigned short* gsrc) {
#if USE_ASYNC
  __builtin_amdgcn_global_load_async_to_lds_b128(
      (GLOBAL_AS v4i*)gsrc, (LDS_AS v4i*)lds, 0, 0);
#else
  *(ushort8*)lds = *(const ushort8*)gsrc;
#endif
}
static __device__ __forceinline__ void async_wait_all() {
#if USE_ASYNC
  __builtin_amdgcn_s_wait_asynccnt(0);
#endif
}

// ---------------- LayerNorm 1 + concat (f32 xn + bf16 xc padded) -------------
__global__ __launch_bounds__(256) void ln_front_kernel(
    const float* __restrict__ x, const float* __restrict__ cond,
    const float* __restrict__ g1, const float* __restrict__ b1,
    float* __restrict__ xn, unsigned short* __restrict__ xc) {
  const int row = blockIdx.x;
  const int tid = threadIdx.x;
  const int lane = tid & 31, wid = tid >> 5;
  const float* xr = x + (size_t)row * EE;
  float v[4], s = 0.f, ss = 0.f;
#pragma unroll
  for (int i = 0; i < 4; i++) { v[i] = xr[tid + i * 256]; s += v[i]; ss += v[i] * v[i]; }
#pragma unroll
  for (int m = 1; m < 32; m <<= 1) { s += __shfl_xor(s, m); ss += __shfl_xor(ss, m); }
  __shared__ float sb[16];
  if (lane == 0) { sb[wid] = s; sb[8 + wid] = ss; }
  __syncthreads();
  float S = 0.f, SQ = 0.f;
#pragma unroll
  for (int w = 0; w < 8; w++) { S += sb[w]; SQ += sb[8 + w]; }
  const float mean = S * (1.f / EE);
  const float var  = SQ * (1.f / EE) - mean * mean;
  const float inv  = rsqrtf(var + 1e-5f);
  unsigned short* xcr = xc + (size_t)row * DP;
#pragma unroll
  for (int i = 0; i < 4; i++) {
    const int c = tid + i * 256;
    const float xv = (v[i] - mean) * inv * g1[c] + b1[c];
    xn[(size_t)row * EE + c] = xv;
    xcr[c] = f2bf(xv);
  }
  if (tid < CSZ)      xcr[EE + tid] = f2bf(cond[(size_t)row * CSZ + tid]);
  else if (tid < 32)  xcr[EE + tid] = 0;   // zero pad cols 1040..1055
}

// ---------------- Final residual + LayerNorm 2 -------------------------------
__global__ __launch_bounds__(256) void ln_back_kernel(
    const float* __restrict__ h, const float* __restrict__ ff,
    const float* __restrict__ g2, const float* __restrict__ b2,
    float* __restrict__ out) {
  const int row = blockIdx.x;
  const int tid = threadIdx.x;
  const int lane = tid & 31, wid = tid >> 5;
  float v[4], s = 0.f, ss = 0.f;
#pragma unroll
  for (int i = 0; i < 4; i++) {
    const int c = tid + i * 256;
    const float t = h[(size_t)row * EE + c] + ff[(size_t)row * EE + c];
    v[i] = t; s += t; ss += t * t;
  }
#pragma unroll
  for (int m = 1; m < 32; m <<= 1) { s += __shfl_xor(s, m); ss += __shfl_xor(ss, m); }
  __shared__ float sb[16];
  if (lane == 0) { sb[wid] = s; sb[8 + wid] = ss; }
  __syncthreads();
  float S = 0.f, SQ = 0.f;
#pragma unroll
  for (int w = 0; w < 8; w++) { S += sb[w]; SQ += sb[8 + w]; }
  const float mean = S * (1.f / EE);
  const float var  = SQ * (1.f / EE) - mean * mean;
  const float inv  = rsqrtf(var + 1e-5f);
#pragma unroll
  for (int i = 0; i < 4; i++) {
    const int c = tid + i * 256;
    out[(size_t)row * EE + c] = (v[i] - mean) * inv * g2[c] + b2[c];
  }
}

// ---------------- Weight f32 -> bf16 with column padding ---------------------
__global__ __launch_bounds__(256) void wcvt_kernel(
    const float* __restrict__ src, unsigned short* __restrict__ dst,
    int R, int C, int Cp) {
  const size_t idx = (size_t)blockIdx.x * 256 + threadIdx.x;
  if (idx >= (size_t)R * Cp) return;
  const size_t r = idx / Cp;
  const int c = (int)(idx - r * Cp);
  dst[idx] = (c < C) ? f2bf(src[r * C + c]) : (unsigned short)0;
}

// ---------------- Extended key weight: [n][c*D+d]=Wk[c,n,d], [n][NC*D+c]=bk --
__global__ __launch_bounds__(256) void wk_build_kernel(
    const float* __restrict__ Wk, const float* __restrict__ bk,
    unsigned short* __restrict__ dst) {
  const size_t idx = (size_t)blockIdx.x * 256 + threadIdx.x;
  if (idx >= (size_t)DD * K2) return;
  const size_t n = idx / K2;
  const int j = (int)(idx - n * K2);
  unsigned short v;
  if (j < NCC * DD) {
    const int c = j / DD, d = j - c * DD;
    v = f2bf(Wk[((size_t)c * DD + n) * DD + d]);
  } else if (j < NCC * DD + NCC) {
    const int c = j - NCC * DD;
    v = f2bf(bk[(size_t)c * DD + n]);
  } else {
    v = 0;
  }
  dst[idx] = v;
}

// ---------------- X2[b,s, c*D+d] = w_c * xc_d ; X2[..., NC*D+c] = w_c --------
__global__ __launch_bounds__(256) void x2_build_kernel(
    const unsigned short* __restrict__ xc, const float* __restrict__ cond,
    unsigned short* __restrict__ X2) {
  const int row = blockIdx.x, tid = threadIdx.x;
  __shared__ float wsh[NCC];
  if (tid < NCC) wsh[tid] = cond[(size_t)row * CSZ + tid];
  __syncthreads();
  const unsigned short* xr = xc + (size_t)row * DP;
  unsigned short* orow = X2 + (size_t)row * K2;
  for (int j = tid; j < NCC * DD; j += 256) {
    const int c = j / DD, d = j - c * DD;
    orow[j] = f2bf(wsh[c] * bf2f(xr[d]));
  }
  if (tid < NCC)     orow[NCC * DD + tid] = f2bf(wsh[tid]);
  else if (tid < 32) orow[NCC * DD + tid] = 0;  // zero pad to K2
}

// ---------------- bf16 WMMA GEMM: C(MxN) = A(MxK) @ W(NxK)^T -----------------
// Block: 256 threads = 8 waves arranged 4(M) x 2(N); wave tile 32x64.
// A/B 128x32 tiles double-buffered in LDS via async load-to-LDS.
// EPI: 0=f32+bias (ff2)  1=relu->bf16 (ff1)  2=(+bias)*scale -> heads bf16 (q,k)
//      3=+bias -> transposed-V bf16        4=+bias + residual -> f32 & bf16 (o)
template <int EPI>
__global__ __launch_bounds__(256) void gemm_bf16_kernel(
    const unsigned short* __restrict__ A, int lda,
    const unsigned short* __restrict__ W, int ldw,
    const float* __restrict__ bias, int Nreal, int K, float scale,
    float* __restrict__ out_f32, unsigned short* __restrict__ out_b16,
    const float* __restrict__ res) {
  __shared__ unsigned short ast[2][128 * 32];
  __shared__ unsigned short bst[2][128 * 32];
  const int tid = threadIdx.x;
  const int lane = tid & 31, wid = tid >> 5;
  const int g = lane >> 4, lr = lane & 15;
  const int wm = wid >> 1, wn = wid & 1;
  const int m0blk = blockIdx.y * 128;
  const int n0blk = blockIdx.x * 128;
  const int m0 = m0blk + wm * 32;
  const int n0 = n0blk + wn * 64;

  v8f acc[2][4];
#pragma unroll
  for (int i = 0; i < 2; i++)
#pragma unroll
    for (int j = 0; j < 4; j++) acc[i][j] = zero8();

  // cooperative stage of one 128x32 A tile + 128x32 B tile (512 x 16B each)
  auto stage = [&](int kk, int buf) {
#pragma unroll
    for (int j = 0; j < 2; j++) {
      const int c = tid + j * 256;          // 0..511
      const int row = c >> 2, seg = c & 3;  // 128 rows x 4 x 16B
      cp16(&ast[buf][row * 32 + seg * 8],
           A + (size_t)(m0blk + row) * lda + kk + seg * 8);
      int nr = n0blk + row;
      if (nr >= Nreal) nr = Nreal - 1;      // clamp; stores guarded in epilogue
      cp16(&bst[buf][row * 32 + seg * 8],
           W + (size_t)nr * ldw + kk + seg * 8);
    }
  };

  const int nk = K >> 5;
  stage(0, 0);
  for (int kc = 0; kc < nk; kc++) {
    async_wait_all();
    __syncthreads();                         // tile kc resident, prev reads done
    if (kc + 1 < nk) stage((kc + 1) * 32, (kc + 1) & 1);
    const unsigned short* as_ = ast[kc & 1];
    const unsigned short* bs_ = bst[kc & 1];
    v16bf af[2], bfr[4];
#pragma unroll
    for (int mt = 0; mt < 2; mt++) {
      const unsigned short* p = as_ + (wm * 32 + mt * 16 + lr) * 32 + g * 8;
      af[mt] = load_frag(p, p + 16);
    }
#pragma unroll
    for (int nt = 0; nt < 4; nt++) {
      const unsigned short* p = bs_ + (wn * 64 + nt * 16 + lr) * 32 + g * 8;
      bfr[nt] = load_frag(p, p + 16);
    }
#pragma unroll
    for (int mt = 0; mt < 2; mt++)
#pragma unroll
      for (int nt = 0; nt < 4; nt++)
        acc[mt][nt] = wmma_bf16(af[mt], bfr[nt], acc[mt][nt]);
  }

#pragma unroll
  for (int mt = 0; mt < 2; mt++) {
#pragma unroll
    for (int nt = 0; nt < 4; nt++) {
      const int ncol = n0 + nt * 16 + lr;
      if (ncol >= Nreal) continue;
      const float bb = bias ? bias[ncol] : 0.f;
#pragma unroll
      for (int r = 0; r < 8; r++) {
        const int m = m0 + mt * 16 + r + 8 * g;     // C layout: row = r + 8*(lane/16)
        float val = acc[mt][nt][r] + bb;
        if (EPI == 0) {
          out_f32[(size_t)m * Nreal + ncol] = val;
        } else if (EPI == 1) {
          val = val > 0.f ? val : 0.f;
          out_b16[(size_t)m * Nreal + ncol] = f2bf(val);
        } else if (EPI == 2) {                       // q / k -> (B,H,S,DHP)
          val *= scale;
          const int b = m >> 11, s = m & (SS - 1);
          const int hh = ncol / DHH, d = ncol - hh * DHH;
          out_b16[((size_t)(b * HH + hh) * SS + s) * DHP + d] = f2bf(val);
        } else if (EPI == 3) {                       // v -> (B,H,DHP,S) transposed
          const int b = m >> 11, s = m & (SS - 1);
          const int hh = ncol / DHH, d = ncol - hh * DHH;
          out_b16[((size_t)(b * HH + hh) * DHP + d) * SS + s] = f2bf(val);
        } else {                                     // o-proj: h = xn + o[:, :E]
          const float hv = val + res[(size_t)m * EE + ncol];
          out_f32[(size_t)m * EE + ncol] = hv;
          out_b16[(size_t)m * EE + ncol] = f2bf(hv);
        }
      }
    }
  }
}

// ---------------- Flash attention -------------------------------------------
// Block = 8 waves, one (b,h), 128 q rows; kv streamed in steps of 32 through
// double-buffered LDS tiles staged with async load-to-LDS (shared by all waves).
__global__ __launch_bounds__(256) void attn_kernel(
    const unsigned short* __restrict__ q,    // (B,H,S,DHP) pre-scaled bf16
    const unsigned short* __restrict__ k,    // (B,H,S,DHP) bf16
    const unsigned short* __restrict__ vt,   // (B,H,DHP,S) bf16
    unsigned short* __restrict__ oatt) {     // (B,S,DP)   bf16
  __shared__ unsigned short kst[2][32 * DHP];   // 2 x 6KB
  __shared__ unsigned short vst[2][DHP * 32];   // 2 x 6KB
  __shared__ unsigned short plds[8 * 16 * 32];  // per-wave 16x32 P tile (8KB)
  const int tid = threadIdx.x;
  const int lane = tid & 31, wid = tid >> 5;
  const int g = lane >> 4, lr = lane & 15;
  const int qt = blockIdx.x & 15;
  const int bh = blockIdx.x >> 4;
  const int b = bh >> 4, h = bh & 15;
  const int s0 = qt * 128 + wid * 16;

  const unsigned short* qb = q  + (size_t)bh * SS * DHP;
  const unsigned short* kb = k  + (size_t)bh * SS * DHP;
  const unsigned short* vb = vt + (size_t)bh * DHP * SS;

  // stage K tile (32 x 96 = 384 chunks) + V tile (96 x 32 = 384 chunks) of 16B
  auto stage = [&](int kv, int buf) {
#pragma unroll
    for (int j = 0; j < 3; j++) {
      const int c = tid + j * 256;
      if (c < 384) {
        const int row = c / 12, seg = c - row * 12;   // 12 x 16B per K row
        cp16(&kst[buf][row * DHP + seg * 8],
             kb + (size_t)(kv + row) * DHP + seg * 8);
      } else {
        const int c2 = c - 384;                       // 2 x 16B per V row
        const int vrow = c2 >> 1, vseg = c2 & 1;
        cp16(&vst[buf][vrow * 32 + vseg * 8],
             vb + (size_t)vrow * SS + kv + vseg * 8);
      }
    }
  };

  v16bf qa[3];
#pragma unroll
  for (int c = 0; c < 3; c++) {
    const unsigned short* p = qb + (size_t)(s0 + lr) * DHP + c * 32 + g * 8;
    qa[c] = load_frag(p, p + 16);
  }
  const v16bf onef = ones16();
  float Mr[8], Lr[8];
  v8f ot[6];
#pragma unroll
  for (int r = 0; r < 8; r++) { Mr[r] = -1e30f; Lr[r] = 0.f; }
#pragma unroll
  for (int t = 0; t < 6; t++) ot[t] = zero8();

  unsigned short* pw = plds + wid * 512;

  stage(0, 0);
  for (int kvi = 0; kvi < SS / 32; kvi++) {
    const int kv = kvi * 32;
    async_wait_all();
    __syncthreads();                          // tile kvi resident, prev reads done
    if (kvi + 1 < SS / 32) stage(kv + 32, (kvi + 1) & 1);
    const unsigned short* ks_ = kst[kvi & 1];
    const unsigned short* vs_ = vst[kvi & 1];

    v8f st[2];
    st[0] = zero8(); st[1] = zero8();
#pragma unroll
    for (int nt = 0; nt < 2; nt++)
#pragma unroll
      for (int c = 0; c < 3; c++) {
        const unsigned short* kp = ks_ + (nt * 16 + lr) * DHP + c * 32 + g * 8;
        st[nt] = wmma_bf16(qa[c], load_frag(kp, kp + 16), st[nt]);
      }
    // online softmax: row max via 16-lane butterfly (matches C-tile row mapping)
    float fac[8];
#pragma unroll
    for (int r = 0; r < 8; r++) {
      float x = fmaxf(st[0][r], st[1][r]);
#pragma unroll
      for (int m = 1; m < 16; m <<= 1) x = fmaxf(x, __shfl_xor(x, m));
      const float mnew = fmaxf(Mr[r], x);
      fac[r] = exp2f((Mr[r] - mnew) * 1.44269504f);
      Mr[r] = mnew;
    }
#pragma unroll
    for (int nt = 0; nt < 2; nt++)
#pragma unroll
      for (int r = 0; r < 8; r++) {
        const float p = exp2f((st[nt][r] - Mr[r]) * 1.44269504f);
        pw[(r + 8 * g) * 32 + nt * 16 + lr] = f2bf(p);  // C-layout -> LDS
      }
#pragma unroll
    for (int t = 0; t < 6; t++)
#pragma unroll
      for (int r = 0; r < 8; r++) ot[t][r] *= fac[r];

    asm volatile("s_wait_dscnt 0x0" ::: "memory");      // per-wave P round trip

    const unsigned short* pr = pw + lr * 32 + g * 8;    // A-layout read of P
    const v16bf pa = load_frag(pr, pr + 16);
    // row sums of P via WMMA against all-ones B (every column = row sum)
    const v8f ls = wmma_bf16(pa, onef, zero8());
#pragma unroll
    for (int r = 0; r < 8; r++) Lr[r] = Lr[r] * fac[r] + ls[r];
#pragma unroll
    for (int t = 0; t < 6; t++) {
      const unsigned short* vp = vs_ + (t * 16 + lr) * 32 + g * 8;
      ot[t] = wmma_bf16(pa, load_frag(vp, vp + 16), ot[t]);
    }
  }
#pragma unroll
  for (int r = 0; r < 8; r++) {
    const float inv = Lr[r] > 0.f ? 1.f / Lr[r] : 0.f;
#pragma unroll
    for (int t = 0; t < 6; t++) ot[t][r] *= inv;
  }
#pragma unroll
  for (int t = 0; t < 6; t++) {
    const int d = t * 16 + lr;
    if (d < DHH) {
#pragma unroll
      for (int r = 0; r < 8; r++) {
        const int srow = s0 + r + 8 * g;
        oatt[((size_t)(b * SS + srow)) * DP + h * DHH + d] = f2bf(ot[t][r]);
      }
    }
  }
  if (h == 0) {  // zero the K-pad columns of oatt for the o-projection GEMM
    for (int idx = tid; idx < 128 * 16; idx += 256) {
      const int rr = idx >> 4, cc = idx & 15;
      oatt[((size_t)(b * SS + qt * 128 + rr)) * DP + DD + cc] = 0;
    }
  }
}

// ---------------------------------------------------------------------------
extern "C" void kernel_launch(void* const* d_in, const int* in_sizes, int n_in,
                              void* d_out, int out_size, void* d_ws, size_t ws_size,
                              hipStream_t stream) {
  (void)in_sizes; (void)n_in; (void)out_size; (void)ws_size;
  const float* x    = (const float*)d_in[0];
  const float* cond = (const float*)d_in[1];
  const float* Wq   = (const float*)d_in[2];
  const float* bq   = (const float*)d_in[3];
  const float* Wv   = (const float*)d_in[4];
  const float* bv   = (const float*)d_in[5];
  const float* Wk   = (const float*)d_in[6];
  const float* bk   = (const float*)d_in[7];
  const float* Wo   = (const float*)d_in[8];
  const float* bo   = (const float*)d_in[9];
  const float* g1   = (const float*)d_in[10];
  const float* b1   = (const float*)d_in[11];
  const float* g2   = (const float*)d_in[12];
  const float* b2   = (const float*)d_in[13];
  const float* Wf1  = (const float*)d_in[14];
  const float* bf1  = (const float*)d_in[15];
  const float* Wf2  = (const float*)d_in[16];
  const float* bf2  = (const float*)d_in[17];
  float* out = (float*)d_out;

  char* base = (char*)d_ws;
  size_t off = 0;
  auto alloc = [&](size_t bytes) -> char* {
    char* p = base + off;
    off = (off + bytes + 255) & ~(size_t)255;
    return p;
  };
  float* xn    = (float*)alloc((size_t)NTOK * EE * 4);
  float* hbuf  = (float*)alloc((size_t)NTOK * EE * 4);
  float* ffo   = (float*)alloc((size_t)NTOK * EE * 4);
  unsigned short* xc   = (unsigned short*)alloc((size_t)NTOK * DP * 2);
  unsigned short* hb   = (unsigned short*)alloc((size_t)NTOK * EE * 2);
  unsigned short* oat  = (unsigned short*)alloc((size_t)NTOK * DP * 2);
  unsigned short* X2   = (unsigned short*)alloc((size_t)NTOK * K2 * 2);
  unsigned short* Wqb  = (unsigned short*)alloc((size_t)DD * DP * 2);
  unsigned short* Wvb  = (unsigned short*)alloc((size_t)DD * DP * 2);
  unsigned short* Wob  = (unsigned short*)alloc((size_t)DD * DP * 2);
  unsigned short* Wkb  = (unsigned short*)alloc((size_t)DD * K2 * 2);
  unsigned short* Wf1b = (unsigned short*)alloc((size_t)DFF * EE * 2);
  unsigned short* Wf2b = (unsigned short*)alloc((size_t)EE * DFF * 2);
  unsigned short* qbuf = (unsigned short*)alloc((size_t)BB * HH * SS * DHP * 2);
  unsigned short* kbuf = (unsigned short*)alloc((size_t)BB * HH * SS * DHP * 2);
  unsigned short* vtb  = (unsigned short*)alloc((size_t)BB * HH * DHP * SS * 2);
  unsigned short* ffh  = (unsigned short*)alloc((size_t)NTOK * DFF * 2);

  // zero pads of q/k/v (q pads MUST be zero; k/v zeroed for determinism)
  (void)hipMemsetAsync(qbuf, 0, (size_t)BB * HH * SS * DHP * 2, stream);
  (void)hipMemsetAsync(kbuf, 0, (size_t)BB * HH * SS * DHP * 2, stream);
  (void)hipMemsetAsync(vtb,  0, (size_t)BB * HH * DHP * SS * 2, stream);

  ln_front_kernel<<<NTOK, 256, 0, stream>>>(x, cond, g1, b1, xn, xc);

  auto cdiv256 = [](size_t t) { return (unsigned)((t + 255) / 256); };
  wcvt_kernel<<<cdiv256((size_t)DD * DP), 256, 0, stream>>>(Wq, Wqb, DD, DD, DP);
  wcvt_kernel<<<cdiv256((size_t)DD * DP), 256, 0, stream>>>(Wv, Wvb, DD, DD, DP);
  wcvt_kernel<<<cdiv256((size_t)DD * DP), 256, 0, stream>>>(Wo, Wob, DD, DD, DP);
  wcvt_kernel<<<cdiv256((size_t)DFF * EE), 256, 0, stream>>>(Wf1, Wf1b, DFF, EE, EE);
  wcvt_kernel<<<cdiv256((size_t)EE * DFF), 256, 0, stream>>>(Wf2, Wf2b, EE, DFF, DFF);
  wk_build_kernel<<<cdiv256((size_t)DD * K2), 256, 0, stream>>>(Wk, bk, Wkb);
  x2_build_kernel<<<NTOK, 256, 0, stream>>>(xc, cond, X2);

  const float qscale = 1.f / sqrtf((float)DHH);
  const dim3 blk(256);
  // q = (xc @ Wq^T + bq) * scale  -> heads layout
  gemm_bf16_kernel<2><<<dim3(9, 64), blk, 0, stream>>>(
      xc, DP, Wqb, DP, bq, DD, DP, qscale, nullptr, qbuf, nullptr);
  // v = xc @ Wv^T + bv            -> transposed heads layout
  gemm_bf16_kernel<3><<<dim3(9, 64), blk, 0, stream>>>(
      xc, DP, Wvb, DP, bv, DD, DP, 1.f, nullptr, vtb, nullptr);
  // k = X2 @ Wk_ext^T (w-mixture + bk folded into K dim) -> heads layout
  gemm_bf16_kernel<2><<<dim3(9, 64), blk, 0, stream>>>(
      X2, K2, Wkb, K2, nullptr, DD, K2, 1.f, nullptr, kbuf, nullptr);
  // attention
  attn_kernel<<<BB * HH * (SS / 128), 256, 0, stream>>>(qbuf, kbuf, vtb, oat);
  // h = xn + (o @ Wo^T + bo)[:, :E]   (f32 + bf16 copy)
  gemm_bf16_kernel<4><<<dim3(8, 64), blk, 0, stream>>>(
      oat, DP, Wob, DP, bo, EE, DP, 1.f, hbuf, hb, xn);
  // ffh = relu(h @ Wf1^T + bf1)
  gemm_bf16_kernel<1><<<dim3(32, 64), blk, 0, stream>>>(
      hb, EE, Wf1b, EE, bf1, DFF, EE, 1.f, nullptr, ffh, nullptr);
  // ffo = ffh @ Wf2^T + bf2
  gemm_bf16_kernel<0><<<dim3(8, 64), blk, 0, stream>>>(
      ffh, DFF, Wf2b, DFF, bf2, EE, DFF, 1.f, ffo, nullptr, nullptr);
  // out = LayerNorm(h + ffo)
  ln_back_kernel<<<NTOK, 256, 0, stream>>>(hbuf, ffo, g2, b2, out);
}
